// Generator_35957466202124
// MI455X (gfx1250) — matrix operand order repaired and checked by reading, not compiled, not run on hardware
//
#include <hip/hip_runtime.h>

// ---------------------------------------------------------------------------
// MI455X (gfx1250) implementation of the per-channel dense + causal-cumsum net.
//
//   layer_first            : int pixels -> relu(w*x+b) -> chunk-local scan
//   scan_chunks            : exclusive prefix of per-chunk sums (carry)
//   3x { layer_dense_scan  : (Xs + carry) -> bf16 WMMA 64x64 GEMM + relu
//                            -> chunk-local scan (in-place on Xs)
//        scan_chunks }
//   layer_final            : (Xs + carry) -> bf16 WMMA 64x(8 pad 16) -> f32 out
//
// Activation tensor is bf16 [B][S][64] (~96MB, L2-resident); all global
// streaming of it is done as packed u32 feature-pairs (128B per wave access).
// ---------------------------------------------------------------------------

typedef __bf16 bf16_t;
typedef __attribute__((ext_vector_type(16))) __bf16 v16bf;
typedef __attribute__((ext_vector_type(8)))  __bf16 v8bf;
typedef __attribute__((ext_vector_type(8)))  float  v8f;

#define NBATCH   4
#define S_TOK    196608              // 65536 pixels * 3 channels per batch
#define T_CHUNK  384                 // tokens per chunk (128 per channel)
#define NCHUNK   512                 // S_TOK / T_CHUNK
#define FILT     64
#define INV_NORM (1.0f / 196608.0f)  // CUMSUM_NORM = 256*256*3

union bfpair { unsigned u; bf16_t h[2]; };

// ---- shared scan: Y[384][64] bf16 in LDS -> scaled inclusive scan ----------
// 256 threads = 32 feature-pairs x 8 segments of 48 tokens.
__device__ __forceinline__ void chunk_scan_write(
    const bf16_t* __restrict__ Y, float* __restrict__ segs,
    bf16_t* __restrict__ Xs, float* __restrict__ Sums,
    size_t tokBase, size_t sumBase)
{
  __syncthreads();                       // Y fully written
  const int tid = threadIdx.x;
  const int fp = tid & 31;               // feature pair -> features 2fp, 2fp+1
  const int g  = tid >> 5;               // segment 0..7 (48 tokens each)
  const unsigned* YU = (const unsigned*)Y;

  float s0 = 0.0f, s1 = 0.0f;
  for (int i = 0; i < 48; ++i) {
    bfpair p; p.u = YU[(g * 48 + i) * 32 + fp];
    s0 += (float)p.h[0]; s1 += (float)p.h[1];
  }
  segs[g * 64 + 2 * fp]     = s0;
  segs[g * 64 + 2 * fp + 1] = s1;
  __syncthreads();

  float r0 = 0.0f, r1 = 0.0f;
  for (int gg = 0; gg < g; ++gg) {
    r0 += segs[gg * 64 + 2 * fp];
    r1 += segs[gg * 64 + 2 * fp + 1];
  }
  unsigned* XoU = (unsigned*)(Xs + tokBase * 64);
  for (int i = 0; i < 48; ++i) {
    bfpair p; p.u = YU[(g * 48 + i) * 32 + fp];
    r0 += (float)p.h[0]; r1 += (float)p.h[1];
    bfpair q;
    q.h[0] = (bf16_t)(r0 * INV_NORM);
    q.h[1] = (bf16_t)(r1 * INV_NORM);
    XoU[(g * 48 + i) * 32 + fp] = q.u;
  }
  if (g == 0) {
    float t0 = 0.0f, t1 = 0.0f;
    for (int gg = 0; gg < 8; ++gg) {
      t0 += segs[gg * 64 + 2 * fp];
      t1 += segs[gg * 64 + 2 * fp + 1];
    }
    Sums[sumBase + 2 * fp]     = t0 * INV_NORM;
    Sums[sumBase + 2 * fp + 1] = t1 * INV_NORM;
  }
}

// ---- chunk input load: (Xs + carry) -> bf16 LDS, packed u32 ----------------
__device__ __forceinline__ void load_chunk_packed(
    const bf16_t* __restrict__ Xs, const float* __restrict__ CarryIn,
    bf16_t* __restrict__ Sin, size_t base64, size_t carryBase)
{
  const int tid = threadIdx.x;
  const int fp0 = tid & 31;              // feature pair
  const int r0  = tid >> 5;              // 0..7
  const float cv0 = CarryIn[carryBase + 2 * fp0];
  const float cv1 = CarryIn[carryBase + 2 * fp0 + 1];
  const unsigned* XsU = (const unsigned*)(Xs + base64);
  unsigned* SinU = (unsigned*)Sin;
  for (int it = 0; it < 48; ++it) {
    int row = r0 + it * 8;
    bfpair p; p.u = XsU[row * 32 + fp0];
    bfpair q;
    q.h[0] = (bf16_t)((float)p.h[0] + cv0);
    q.h[1] = (bf16_t)((float)p.h[1] + cv1);
    SinU[row * 32 + fp0] = q.u;
  }
}

// ---- layer 0: din=1 outer product + relu + scan ----------------------------
__global__ __launch_bounds__(256) void layer_first(
    const int* __restrict__ Ex, const float* __restrict__ W0,
    const float* __restrict__ B0, bf16_t* __restrict__ Xs,
    float* __restrict__ Sums)
{
  extern __shared__ char smem[];
  bf16_t* Y    = (bf16_t*)smem;          // 384*64 bf16
  float*  w0   = (float*)(Y + 24576);    // 192
  float*  b0   = w0 + 192;               // 192
  float*  segs = b0 + 192;               // 512

  const int tid = threadIdx.x;
  const int k = blockIdx.x, b = blockIdx.y;
  const size_t tokBase = (size_t)b * S_TOK + (size_t)k * T_CHUNK;

  if (tid < 192) { w0[tid] = W0[tid]; b0[tid] = B0[tid]; }
  __syncthreads();

  const int fp0 = tid & 31, r0 = tid >> 5;
  unsigned* YU = (unsigned*)Y;
  for (int it = 0; it < 48; ++it) {
    int row = r0 + it * 8;               // 0..383, chunk base divisible by 3
    int c = row % 3;
    float x = (float)Ex[tokBase + row] * 0.25f - 1.0f;  // v/8*2-1
    bfpair q;
    q.h[0] = (bf16_t)fmaxf(x * w0[c * 64 + 2 * fp0]     + b0[c * 64 + 2 * fp0],     0.0f);
    q.h[1] = (bf16_t)fmaxf(x * w0[c * 64 + 2 * fp0 + 1] + b0[c * 64 + 2 * fp0 + 1], 0.0f);
    YU[row * 32 + fp0] = q.u;
  }
  chunk_scan_write(Y, segs, Xs, Sums, tokBase, ((size_t)b * NCHUNK + k) * 64);
}

// ---- carry scan: exclusive prefix of per-chunk sums per feature ------------
__global__ __launch_bounds__(256) void scan_chunks(
    const float* __restrict__ Sums, float* __restrict__ Carry)
{
  extern __shared__ char smem[];
  float* lds = (float*)smem;             // 512*64 floats = 128KB
  const int b = blockIdx.x, tid = threadIdx.x;
  const size_t base = (size_t)b * (NCHUNK * 64);
  for (int i = tid; i < NCHUNK * 64; i += 256) lds[i] = Sums[base + i];
  __syncthreads();
  if (tid < 64) {
    float run = 0.0f;
    for (int kk = 0; kk < NCHUNK; ++kk) {
      float v = lds[kk * 64 + tid];
      Carry[base + (size_t)kk * 64 + tid] = run;
      run += v;
    }
  }
}

// ---- 64->64 dense (bf16 WMMA) + relu + scan, in-place on Xs ----------------
__global__ __launch_bounds__(256) void layer_dense_scan(
    bf16_t* __restrict__ Xs,             // [B][S][64] in/out (chunk-local safe)
    const float* __restrict__ CarryIn,   // [B][512][64]
    const float* __restrict__ Wg,        // [3][64][64]  (c, din, dout)
    const float* __restrict__ Bg,        // [3][64]
    float* __restrict__ Sums)            // [B][512][64]
{
  extern __shared__ char smem[];
  bf16_t* Sin   = (bf16_t*)smem;          // 24576 halfs
  bf16_t* Wt    = Sin + 24576;            // 12288 halfs: [c][dout][din]
  float*  biasL = (float*)(Wt + 12288);   // 192
  float*  segs  = biasL + 192;            // 512
  bf16_t* Y     = (bf16_t*)(segs + 512);  // 24576 halfs

  const int tid = threadIdx.x;
  const int k = blockIdx.x, b = blockIdx.y;
  const size_t tokBase = (size_t)b * S_TOK + (size_t)k * T_CHUNK;

  load_chunk_packed(Xs, CarryIn, Sin, tokBase * 64, ((size_t)b * NCHUNK + k) * 64);

  // weights transposed: Wt[c][e][d] = Wg[c][d][e]
  for (int idx = tid; idx < 3 * 64 * 64; idx += 256) {
    int c = idx >> 12, r = idx & 4095, d = r >> 6, e = r & 63;
    Wt[(c * 64 + e) * 64 + d] = (bf16_t)Wg[idx];
  }
  if (tid < 192) biasL[tid] = Bg[tid];
  __syncthreads();

  // 3 channels x 8 M-tiles x 4 N-tiles = 96 output tiles, 8 waves
  const int lane = tid & 31, wave = tid >> 5;
  const int col = lane & 15, hi16 = lane >> 4;
  for (int tile = wave; tile < 96; tile += 8) {
    const int c = tile >> 5, nt = tile & 3, mt = (tile >> 2) & 7;
    const float bv = biasL[c * 64 + nt * 16 + col];
    v8f acc;
#pragma unroll
    for (int i = 0; i < 8; ++i) acc[i] = bv;
#pragma unroll
    for (int ks = 0; ks < 2; ++ks) {
      // A 16x32 bf16: lanes 0-15 row=lane K 0-7/16-23; lanes 16-31 K 8-15/24-31
      int j = 3 * (mt * 16 + col) + c;
      const bf16_t* ap = &Sin[j * 64 + ks * 32 + (hi16 ? 8 : 0)];
      v8bf a0 = *(const v8bf*)ap;
      v8bf a1 = *(const v8bf*)(ap + 16);
      v16bf A = __builtin_shufflevector(a0, a1, 0,1,2,3,4,5,6,7,8,9,10,11,12,13,14,15);
      // B 32x16 bf16: lane col = lane&15; lanes 0-15 K 0-7/16-23, lanes 16-31 K 8-15/24-31
      const bf16_t* bp = &Wt[(c * 64 + nt * 16 + col) * 64 + ks * 32 + (hi16 ? 8 : 0)];
      v8bf q0 = *(const v8bf*)bp;
      v8bf q1 = *(const v8bf*)(bp + 16);
      v16bf Bm = __builtin_shufflevector(q0, q1, 0,1,2,3,4,5,6,7,8,9,10,11,12,13,14,15);
      acc = __builtin_amdgcn_wmma_f32_16x16x32_bf16(false, A, false, Bm,
                                                    (short)0, acc, false, false);
    }
    // C/D: VGPR r -> row r (+8 for lanes 16-31); relu and scatter into Y
    const int mrow = mt * 16 + hi16 * 8;
#pragma unroll
    for (int r = 0; r < 8; ++r) {
      int j = 3 * (mrow + r) + c;
      Y[j * 64 + nt * 16 + col] = (bf16_t)fmaxf(acc[r], 0.0f);
    }
  }

  chunk_scan_write(Y, segs, Xs, Sums, tokBase, ((size_t)b * NCHUNK + k) * 64);
}

// ---- final 64->8 dense (WMMA, N padded to 16), f32 output ------------------
__global__ __launch_bounds__(256) void layer_final(
    const bf16_t* __restrict__ Xs, const float* __restrict__ CarryIn,
    const float* __restrict__ Wf,    // [3][64][8]
    const float* __restrict__ Bf,    // [3][8]
    float* __restrict__ Out)         // [B][S][8] f32
{
  extern __shared__ char smem[];
  bf16_t* Sin   = (bf16_t*)smem;          // 24576 halfs
  bf16_t* WtF   = Sin + 24576;            // [3][16][64] = 3072 halfs
  float*  biasF = (float*)(WtF + 3072);   // 48

  const int tid = threadIdx.x;
  const int k = blockIdx.x, b = blockIdx.y;
  const size_t tokBase = (size_t)b * S_TOK + (size_t)k * T_CHUNK;

  load_chunk_packed(Xs, CarryIn, Sin, tokBase * 64, ((size_t)b * NCHUNK + k) * 64);

  for (int idx = tid; idx < 3 * 16 * 64; idx += 256) {
    int c = idx >> 10, r = idx & 1023, e = r >> 6, d = r & 63;
    WtF[idx] = (e < 8) ? (bf16_t)Wf[c * 512 + d * 8 + e] : (bf16_t)0.0f;
  }
  if (tid < 48) {
    int c = tid >> 4, e = tid & 15;
    biasF[tid] = (e < 8) ? Bf[c * 8 + e] : 0.0f;
  }
  __syncthreads();

  const int lane = tid & 31, wave = tid >> 5;
  const int col = lane & 15, hi16 = lane >> 4;
  for (int tile = wave; tile < 24; tile += 8) {   // 3 channels x 8 M-tiles
    const int c = tile >> 3, mt = tile & 7;
    const float bv = biasF[c * 16 + col];
    v8f acc;
#pragma unroll
    for (int i = 0; i < 8; ++i) acc[i] = bv;
#pragma unroll
    for (int ks = 0; ks < 2; ++ks) {
      int j = 3 * (mt * 16 + col) + c;
      const bf16_t* ap = &Sin[j * 64 + ks * 32 + (hi16 ? 8 : 0)];
      v8bf a0 = *(const v8bf*)ap;
      v8bf a1 = *(const v8bf*)(ap + 16);
      v16bf A = __builtin_shufflevector(a0, a1, 0,1,2,3,4,5,6,7,8,9,10,11,12,13,14,15);
      const bf16_t* bp = &WtF[(c * 16 + col) * 64 + ks * 32 + (hi16 ? 8 : 0)];
      v8bf q0 = *(const v8bf*)bp;
      v8bf q1 = *(const v8bf*)(bp + 16);
      v16bf Bm = __builtin_shufflevector(q0, q1, 0,1,2,3,4,5,6,7,8,9,10,11,12,13,14,15);
      acc = __builtin_amdgcn_wmma_f32_16x16x32_bf16(false, A, false, Bm,
                                                    (short)0, acc, false, false);
    }
    if (col < 8) {
      const int mrow = mt * 16 + hi16 * 8;
#pragma unroll
      for (int r = 0; r < 8; ++r) {
        int j = 3 * (mrow + r) + c;
        Out[(tokBase + (size_t)j) * 8 + col] = acc[r];
      }
    }
  }
}

// ---------------------------------------------------------------------------
extern "C" void kernel_launch(void* const* d_in, const int* in_sizes, int n_in,
                              void* d_out, int out_size, void* d_ws, size_t ws_size,
                              hipStream_t stream) {
  (void)in_sizes; (void)n_in; (void)out_size; (void)ws_size;
  const int*   Ex = (const int*)d_in[0];    // [4,256,256,3] int32
  const float* W0 = (const float*)d_in[1];  // [3,1,64]
  const float* B0 = (const float*)d_in[2];  // [3,64]
  const float* Wr = (const float*)d_in[3];  // [3,3,64,64]
  const float* Br = (const float*)d_in[4];  // [3,3,64]
  const float* Wf = (const float*)d_in[5];  // [3,64,8]
  const float* Bf = (const float*)d_in[6];  // [3,8]
  float* Out = (float*)d_out;

  char* ws = (char*)d_ws;
  const size_t XS_BYTES  = (size_t)NBATCH * S_TOK * FILT * sizeof(bf16_t); // ~96MB
  const size_t SUM_BYTES = (size_t)NBATCH * NCHUNK * FILT * sizeof(float); // 512KB
  bf16_t* Xs    = (bf16_t*)ws;
  float*  Sums  = (float*)(ws + XS_BYTES);
  float*  Carry = (float*)(ws + XS_BYTES + SUM_BYTES);

  const dim3 grid(NCHUNK, NBATCH), blk(256);
  const size_t sm_first = 24576 * 2 + (192 + 192 + 512) * 4;                  // ~52.7KB
  const size_t sm_layer = (24576 + 12288) * 2 + (192 + 512) * 4 + 24576 * 2;  // ~126KB
  const size_t sm_final = (24576 + 3072) * 2 + 48 * 4;                        // ~55.5KB
  const size_t sm_scan  = (size_t)NCHUNK * FILT * 4;                          // 128KB

  layer_first<<<grid, blk, sm_first, stream>>>(Ex, W0, B0, Xs, Sums);
  scan_chunks<<<dim3(NBATCH), blk, sm_scan, stream>>>(Sums, Carry);
  for (int l = 0; l < 3; ++l) {
    layer_dense_scan<<<grid, blk, sm_layer, stream>>>(
        Xs, Carry, Wr + (size_t)l * 3 * 64 * 64, Br + (size_t)l * 3 * 64, Sums);
    scan_chunks<<<dim3(NBATCH), blk, sm_scan, stream>>>(Sums, Carry);
  }
  layer_final<<<grid, blk, sm_final, stream>>>(Xs, Carry, Wf, Bf, Out);
}
